// MemoryDefense_53927609368952
// MI455X (gfx1250) — matrix operand order repaired and checked by reading, compiled, not run on hardware
//
#include <hip/hip_runtime.h>
#include <hip/hip_bf16.h>

// ---------------------------------------------------------------------------
// MemoryDefense forward, MI455X (gfx1250, wave32, WMMA).
// Big GEMMs (zn@mn^T, w_t@memory, w_nt@memory) use v_wmma_f32_16x16x32_f16
// with 2x2 register blocking (2 b128 loads per WMMA).
// ---------------------------------------------------------------------------

typedef __attribute__((ext_vector_type(16))) _Float16 v16h;
typedef __attribute__((ext_vector_type(8)))  _Float16 v8h;
typedef __attribute__((ext_vector_type(8)))  float    v8f;

#define BATCH 1024
#define FEAT  1024
#define MEM   10000
#define MEMP  10016   // MEM padded to multiple of 32 for WMMA K / N tiling

__device__ __forceinline__ float leaky025(float x) { return x > 0.f ? x : 0.25f * x; }
__device__ __forceinline__ float bnscale(float g)  { return g * rsqrtf(1.0f + 1e-5f); }

__device__ __forceinline__ float block_reduce_sum(float v, float* red) {
  int tid = threadIdx.x;
  red[tid] = v; __syncthreads();
  for (int o = 128; o > 0; o >>= 1) { if (tid < o) red[tid] += red[tid + o]; __syncthreads(); }
  float r = red[0]; __syncthreads();
  return r;
}
__device__ __forceinline__ float block_reduce_max(float v, float* red) {
  int tid = threadIdx.x;
  red[tid] = v; __syncthreads();
  for (int o = 128; o > 0; o >>= 1) { if (tid < o) red[tid] = fmaxf(red[tid], red[tid + o]); __syncthreads(); }
  float r = red[0]; __syncthreads();
  return r;
}

// ---------------- Encoder ----------------
// conv1: (B,1,28,28) -> (B,16,15,15), 1x1 s2 p1, BN eval, leaky 0.25
__global__ __launch_bounds__(256) void enc1_kernel(
    const float* __restrict__ x, const float* __restrict__ w, const float* __restrict__ b,
    const float* __restrict__ g, const float* __restrict__ be, float* __restrict__ h1) {
  int idx = blockIdx.x * 256 + threadIdx.x;
  if (idx >= BATCH * 16 * 15 * 15) return;
  int px = idx % 15, py = (idx / 15) % 15, c = (idx / 225) % 16, bb = idx / 3600;
  int iy = 2 * py - 1, ix = 2 * px - 1;
  float v = 0.f;
  if (iy >= 0 && ix >= 0) v = x[bb * 784 + iy * 28 + ix];
  float t = w[c] * v + b[c];
  t = t * bnscale(g[c]) + be[c];
  h1[idx] = leaky025(t);
}

// conv2: (B,16,15,15) -> (B,32,8,8), 3x3 s2 p1, BN, leaky
__global__ __launch_bounds__(256) void enc2_kernel(
    const float* __restrict__ h1, const float* __restrict__ w, const float* __restrict__ b,
    const float* __restrict__ g, const float* __restrict__ be, float* __restrict__ h2) {
  int idx = blockIdx.x * 256 + threadIdx.x;
  if (idx >= BATCH * 32 * 64) return;
  int ox = idx & 7, oy = (idx >> 3) & 7, oc = (idx >> 6) & 31, bb = idx >> 11;
  float acc = b[oc];
  const float* wp = w + oc * 144;
  const float* ip = h1 + bb * 16 * 225;
  for (int ky = 0; ky < 3; ++ky) {
    int iy = 2 * oy - 1 + ky; if (iy < 0 || iy >= 15) continue;
    for (int kx = 0; kx < 3; ++kx) {
      int ix = 2 * ox - 1 + kx; if (ix < 0 || ix >= 15) continue;
      for (int ic = 0; ic < 16; ++ic)
        acc += ip[ic * 225 + iy * 15 + ix] * wp[ic * 9 + ky * 3 + kx];
    }
  }
  float t = acc * bnscale(g[oc]) + be[oc];
  h2[idx] = leaky025(t);
}

// conv3: (B,32,8,8) -> (B,64,4,4), 3x3 s2 p1, BN, sigmoid -> z (B,1024)
__global__ __launch_bounds__(256) void enc3_kernel(
    const float* __restrict__ h2, const float* __restrict__ w, const float* __restrict__ b,
    const float* __restrict__ g, const float* __restrict__ be, float* __restrict__ z) {
  int idx = blockIdx.x * 256 + threadIdx.x;
  if (idx >= BATCH * 1024) return;
  int ox = idx & 3, oy = (idx >> 2) & 3, oc = (idx >> 4) & 63, bb = idx >> 10;
  float acc = b[oc];
  const float* wp = w + oc * 288;
  const float* ip = h2 + bb * 32 * 64;
  for (int ky = 0; ky < 3; ++ky) {
    int iy = 2 * oy - 1 + ky; if (iy < 0 || iy >= 8) continue;
    for (int kx = 0; kx < 3; ++kx) {
      int ix = 2 * ox - 1 + kx; if (ix < 0 || ix >= 8) continue;
      for (int ic = 0; ic < 32; ++ic)
        acc += ip[ic * 64 + iy * 8 + ix] * wp[ic * 9 + ky * 3 + kx];
    }
  }
  float t = acc * bnscale(g[oc]) + be[oc];
  z[idx] = 1.0f / (1.0f + __expf(-t));
}

// ---------------- Memory prep ----------------
// L2-normalize each 1024-elem row (eps 1e-8), emit f16. One block per row.
// Rows >= nvalid are zero-filled (N/K padding for the WMMA tiling).
__global__ __launch_bounds__(256) void rownorm_f16_kernel(
    const float* __restrict__ src, _Float16* __restrict__ dst, int nvalid) {
  __shared__ float red[256];
  int r = blockIdx.x, tid = threadIdx.x;
  if (r >= nvalid) {
    for (int i = tid; i < FEAT; i += 256) dst[(size_t)r * FEAT + i] = (_Float16)0.f;
    return;
  }
  const float* row = src + (size_t)r * FEAT;
  float s = 0.f;
  for (int i = tid; i < FEAT; i += 256) { float v = row[i]; s += v * v; }
  s = block_reduce_sum(s, red);
  float inv = 1.0f / fmaxf(sqrtf(s), 1e-8f);
  for (int i = tid; i < FEAT; i += 256)
    dst[(size_t)r * FEAT + i] = (_Float16)(row[i] * inv);
}

// memT[f][r] = memory[r][f] as f16, K-padded with zeros to MEMP.
__global__ __launch_bounds__(256) void memT_kernel(
    const float* __restrict__ mem, _Float16* __restrict__ mT) {
  int idx = blockIdx.x * 256 + threadIdx.x;
  if (idx >= FEAT * MEMP) return;
  int r = idx % MEMP, f = idx / MEMP;
  mT[idx] = (r < MEM) ? (_Float16)mem[(size_t)r * FEAT + f] : (_Float16)0.f;
}

// ---------- WMMA GEMM (2x2 tiles/wave): C[M x N] = A[M x K] * Bt[N x K]^T ----------
// Each wave produces a 32x32 C block (4 accumulators); A/B fragments are each
// reused twice -> 2 global b128 loads per v_wmma. 8 waves / 256-thread block.
// Fragment layouts per CDNA5 ISA 7.12.2 (wave32, 16-bit operands).
// M and ldc*16 must be multiples of 32 tiles-wise; N tile-pairs guarded by ntg.
__global__ __launch_bounds__(256) void gemm_nt_wmma2x2_kernel(
    const _Float16* __restrict__ A, const _Float16* __restrict__ Bt,
    float* __restrict__ C, int K, int lda, int ldb, int ldc, int ntg) {
  int wave = threadIdx.x >> 5;
  int lane = threadIdx.x & 31;
  int gx = blockIdx.x * 8 + wave;
  if (gx >= ntg) return;                 // wave-uniform: EXEC stays all-ones
  int m0 = blockIdx.y << 5;              // 32 rows per wave
  int n0 = gx << 5;                      // 32 cols per wave
  int rsel = lane & 15;
  int hs = lane >> 4;
  // A: lane holds row; K halves {k+8*hs+0..7, k+8*hs+16..23}
  const _Float16* Ap0 = A + (size_t)(m0 + rsel) * lda + hs * 8;
  const _Float16* Ap1 = Ap0 + (size_t)16 * lda;
  // B: lane holds col; K halves {k+8*hs+0..7, k+8*hs+16..23}
  const _Float16* Bp0 = Bt + (size_t)(n0 + rsel) * ldb + hs * 8;
  const _Float16* Bp1 = Bp0 + (size_t)16 * ldb;
  v8f acc00 = {0.f,0.f,0.f,0.f,0.f,0.f,0.f,0.f};
  v8f acc01 = acc00, acc10 = acc00, acc11 = acc00;
  for (int k = 0; k < K; k += 32) {
    union { v16h v; v8h h[2]; } a0, a1, b0, b1;
    a0.h[0] = *(const v8h*)(Ap0 + k); a0.h[1] = *(const v8h*)(Ap0 + k + 16);
    a1.h[0] = *(const v8h*)(Ap1 + k); a1.h[1] = *(const v8h*)(Ap1 + k + 16);
    b0.h[0] = *(const v8h*)(Bp0 + k); b0.h[1] = *(const v8h*)(Bp0 + k + 16);
    b1.h[0] = *(const v8h*)(Bp1 + k); b1.h[1] = *(const v8h*)(Bp1 + k + 16);
    acc00 = __builtin_amdgcn_wmma_f32_16x16x32_f16(false, a0.v, false, b0.v, (short)0, acc00, false, false);
    acc01 = __builtin_amdgcn_wmma_f32_16x16x32_f16(false, a0.v, false, b1.v, (short)0, acc01, false, false);
    acc10 = __builtin_amdgcn_wmma_f32_16x16x32_f16(false, a1.v, false, b0.v, (short)0, acc10, false, false);
    acc11 = __builtin_amdgcn_wmma_f32_16x16x32_f16(false, a1.v, false, b1.v, (short)0, acc11, false, false);
  }
  // C: VGPR r -> row (+ 8*hs + r), col (+ rsel)
#pragma unroll
  for (int r = 0; r < 8; ++r) {
    size_t row0 = (size_t)(m0 + hs * 8 + r);
    size_t row1 = row0 + 16;
    C[row0 * ldc + n0 + rsel]      = acc00[r];
    C[row0 * ldc + n0 + 16 + rsel] = acc01[r];
    C[row1 * ldc + n0 + rsel]      = acc10[r];
    C[row1 * ldc + n0 + 16 + rsel] = acc11[r];
  }
}

// ---------------- Softmax + label mask + hardshrink + L1 renorm ----------------
// S has row stride MEMP (padded cols are zeros, never read).
__global__ __launch_bounds__(256) void addr_kernel(
    const float* __restrict__ S, const int* __restrict__ labels,
    float* __restrict__ wt_out, float* __restrict__ wnt_out,
    _Float16* __restrict__ wt_h, _Float16* __restrict__ wnt_h) {
  __shared__ float red[256];
  const int r = blockIdx.x, tid = threadIdx.x;
  const float* s = S + (size_t)r * MEMP;
  float m = -3.402823e38f;
  for (int i = tid; i < MEM; i += 256) m = fmaxf(m, s[i]);
  m = block_reduce_max(m, red);
  float sum = 0.f;
  for (int i = tid; i < MEM; i += 256) sum += __expf(s[i] - m);
  sum = block_reduce_sum(sum, red);
  const float inv = 1.0f / sum;
  const int lab = labels[r];
  const int lo = lab * (MEM / 10), hi = lo + (MEM / 10);
  float ts = 0.f, ns = 0.f;
  for (int i = tid; i < MEM; i += 256) {
    float p = __expf(s[i] - m) * inv;
    float v = (p > 1e-12f) ? p : 0.f;          // hardshrink(lambda=1e-12)
    if (i >= lo && i < hi) ts += v; else ns += v;
  }
  ts = block_reduce_sum(ts, red);
  ns = block_reduce_sum(ns, red);
  const float it = 1.0f / ts, inn = 1.0f / ns;
  const size_t ro = (size_t)r * MEM, rp = (size_t)r * MEMP;
  for (int i = tid; i < MEM; i += 256) {
    float p = __expf(s[i] - m) * inv;
    bool keep = p > 1e-12f;
    bool inb = (i >= lo) && (i < hi);
    float vt = (keep && inb)  ? p * it  : 0.f;
    float vn = (keep && !inb) ? p * inn : 0.f;
    wt_out[ro + i] = vt;
    wnt_out[ro + i] = vn;
    wt_h[rp + i]  = (_Float16)vt;
    wnt_h[rp + i] = (_Float16)vn;
  }
  if (tid < MEMP - MEM) {                       // zero K-pad columns
    wt_h[rp + MEM + tid]  = (_Float16)0.f;
    wnt_h[rp + MEM + tid] = (_Float16)0.f;
  }
}

// ---------------- Decoder (ConvTranspose2d chain) ----------------
// dec1: (B,64,4,4) -> (B,32,7,7), k3 s2 p1 op0, BN, leaky. w:(64,32,3,3)
__global__ __launch_bounds__(256) void dec1_kernel(
    const float* __restrict__ zz, const float* __restrict__ w, const float* __restrict__ b,
    const float* __restrict__ g, const float* __restrict__ be, float* __restrict__ out) {
  int idx = blockIdx.x * 256 + threadIdx.x;
  if (idx >= BATCH * 32 * 49) return;
  int ox = idx % 7, oy = (idx / 7) % 7, oc = (idx / 49) % 32, bb = idx / (32 * 49);
  float acc = b[oc];
  const float* ip = zz + (size_t)bb * 1024;
  for (int ky = 0; ky < 3; ++ky) {
    int ty = oy + 1 - ky; if (ty < 0 || (ty & 1)) continue; int iy = ty >> 1; if (iy >= 4) continue;
    for (int kx = 0; kx < 3; ++kx) {
      int tx = ox + 1 - kx; if (tx < 0 || (tx & 1)) continue; int ix = tx >> 1; if (ix >= 4) continue;
      for (int ic = 0; ic < 64; ++ic)
        acc += ip[ic * 16 + iy * 4 + ix] * w[ic * 288 + oc * 9 + ky * 3 + kx];
    }
  }
  float t = acc * bnscale(g[oc]) + be[oc];
  out[idx] = leaky025(t);
}

// dec2: (B,32,7,7) -> (B,16,14,14), k3 s2 p1 op1, BN, leaky. w:(32,16,3,3)
__global__ __launch_bounds__(256) void dec2_kernel(
    const float* __restrict__ in, const float* __restrict__ w, const float* __restrict__ b,
    const float* __restrict__ g, const float* __restrict__ be, float* __restrict__ out) {
  int idx = blockIdx.x * 256 + threadIdx.x;
  if (idx >= BATCH * 16 * 196) return;
  int ox = idx % 14, oy = (idx / 14) % 14, oc = (idx / 196) % 16, bb = idx / (16 * 196);
  float acc = b[oc];
  const float* ip = in + (size_t)bb * 32 * 49;
  for (int ky = 0; ky < 3; ++ky) {
    int ty = oy + 1 - ky; if (ty < 0 || (ty & 1)) continue; int iy = ty >> 1; if (iy >= 7) continue;
    for (int kx = 0; kx < 3; ++kx) {
      int tx = ox + 1 - kx; if (tx < 0 || (tx & 1)) continue; int ix = tx >> 1; if (ix >= 7) continue;
      for (int ic = 0; ic < 32; ++ic)
        acc += ip[ic * 49 + iy * 7 + ix] * w[ic * 144 + oc * 9 + ky * 3 + kx];
    }
  }
  float t = acc * bnscale(g[oc]) + be[oc];
  out[idx] = leaky025(t);
}

// dec3: (B,16,14,14) -> (B,1,28,28), k3 s2 p1 op1 (no BN/act). w:(16,1,3,3)
__global__ __launch_bounds__(256) void dec3_kernel(
    const float* __restrict__ in, const float* __restrict__ w, const float* __restrict__ b,
    float* __restrict__ out) {
  int idx = blockIdx.x * 256 + threadIdx.x;
  if (idx >= BATCH * 784) return;
  int ox = idx % 28, oy = (idx / 28) % 28, bb = idx / 784;
  float acc = b[0];
  const float* ip = in + (size_t)bb * 16 * 196;
  for (int ky = 0; ky < 3; ++ky) {
    int ty = oy + 1 - ky; if (ty < 0 || (ty & 1)) continue; int iy = ty >> 1; if (iy >= 14) continue;
    for (int kx = 0; kx < 3; ++kx) {
      int tx = ox + 1 - kx; if (tx < 0 || (tx & 1)) continue; int ix = tx >> 1; if (ix >= 14) continue;
      for (int ic = 0; ic < 16; ++ic)
        acc += ip[ic * 196 + iy * 14 + ix] * w[ic * 9 + ky * 3 + kx];
    }
  }
  out[idx] = acc;
}

// ---------------- Classifier ----------------
// cls1: (B,1,28,28) --5x5 conv-> 24x24 --pool2--> (B,6,12,12), relu
__global__ __launch_bounds__(256) void cls1_kernel(
    const float* __restrict__ rec, const float* __restrict__ w, const float* __restrict__ b,
    float* __restrict__ out) {
  int idx = blockIdx.x * 256 + threadIdx.x;
  if (idx >= BATCH * 6 * 144) return;
  int px = idx % 12, py = (idx / 12) % 12, oc = (idx / 144) % 6, bb = idx / 864;
  const float* ip = rec + (size_t)bb * 784;
  const float* wp = w + oc * 25;
  float mx = -3.402823e38f;
  for (int dy = 0; dy < 2; ++dy)
    for (int dx = 0; dx < 2; ++dx) {
      int oy = 2 * py + dy, ox = 2 * px + dx;
      float a = b[oc];
      for (int ky = 0; ky < 5; ++ky)
        for (int kx = 0; kx < 5; ++kx)
          a += ip[(oy + ky) * 28 + ox + kx] * wp[ky * 5 + kx];
      mx = fmaxf(mx, a);
    }
  out[idx] = fmaxf(mx, 0.f);
}

// cls2: (B,6,12,12) --5x5-> 8x8 --pool2--> (B,16,4,4), relu -> flat (B,256)
__global__ __launch_bounds__(256) void cls2_kernel(
    const float* __restrict__ c1, const float* __restrict__ w, const float* __restrict__ b,
    float* __restrict__ out) {
  int idx = blockIdx.x * 256 + threadIdx.x;
  if (idx >= BATCH * 256) return;
  int px = idx & 3, py = (idx >> 2) & 3, oc = (idx >> 4) & 15, bb = idx >> 8;
  const float* ip = c1 + (size_t)bb * 864;
  const float* wp = w + oc * 150;
  float mx = -3.402823e38f;
  for (int dy = 0; dy < 2; ++dy)
    for (int dx = 0; dx < 2; ++dx) {
      int oy = 2 * py + dy, ox = 2 * px + dx;
      float a = b[oc];
      for (int ic = 0; ic < 6; ++ic)
        for (int ky = 0; ky < 5; ++ky)
          for (int kx = 0; kx < 5; ++kx)
            a += ip[ic * 144 + (oy + ky) * 12 + ox + kx] * wp[ic * 25 + ky * 5 + kx];
      mx = fmaxf(mx, a);
    }
  out[idx] = fmaxf(mx, 0.f);
}

// fc: out[b][o] = relu?(in[b] . w[o] + bias[o])
__global__ __launch_bounds__(256) void fc_kernel(
    const float* __restrict__ in, const float* __restrict__ w, const float* __restrict__ b,
    float* __restrict__ out, int indim, int outdim, int do_relu) {
  int idx = blockIdx.x * 256 + threadIdx.x;
  if (idx >= BATCH * outdim) return;
  int o = idx % outdim, bb = idx / outdim;
  const float* ip = in + (size_t)bb * indim;
  const float* wp = w + (size_t)o * indim;
  float a = b[o];
  for (int i = 0; i < indim; ++i) a += ip[i] * wp[i];
  out[idx] = do_relu ? fmaxf(a, 0.f) : a;
}

// ---------------------------------------------------------------------------
extern "C" void kernel_launch(void* const* d_in, const int* in_sizes, int n_in,
                              void* d_out, int out_size, void* d_ws, size_t ws_size,
                              hipStream_t stream) {
  (void)in_sizes; (void)n_in; (void)out_size; (void)ws_size;
  const float* x      = (const float*)d_in[0];
  const int*   labels = (const int*)d_in[1];
  const float* enc_w1 = (const float*)d_in[2];
  const float* enc_b1 = (const float*)d_in[3];
  const float* enc_g1 = (const float*)d_in[4];
  const float* enc_be1= (const float*)d_in[5];
  const float* enc_w2 = (const float*)d_in[6];
  const float* enc_b2 = (const float*)d_in[7];
  const float* enc_g2 = (const float*)d_in[8];
  const float* enc_be2= (const float*)d_in[9];
  const float* enc_w3 = (const float*)d_in[10];
  const float* enc_b3 = (const float*)d_in[11];
  const float* enc_g3 = (const float*)d_in[12];
  const float* enc_be3= (const float*)d_in[13];
  const float* memory = (const float*)d_in[14];
  const float* dec_w1 = (const float*)d_in[15];
  const float* dec_b1 = (const float*)d_in[16];
  const float* dec_g1 = (const float*)d_in[17];
  const float* dec_be1= (const float*)d_in[18];
  const float* dec_w2 = (const float*)d_in[19];
  const float* dec_b2 = (const float*)d_in[20];
  const float* dec_g2 = (const float*)d_in[21];
  const float* dec_be2= (const float*)d_in[22];
  const float* dec_w3 = (const float*)d_in[23];
  const float* dec_b3 = (const float*)d_in[24];
  const float* cls_w1 = (const float*)d_in[25];
  const float* cls_b1 = (const float*)d_in[26];
  const float* cls_w2 = (const float*)d_in[27];
  const float* cls_b2 = (const float*)d_in[28];
  const float* fc1_w  = (const float*)d_in[29];
  const float* fc1_b  = (const float*)d_in[30];
  const float* fc2_w  = (const float*)d_in[31];
  const float* fc2_b  = (const float*)d_in[32];
  const float* fc3_w  = (const float*)d_in[33];
  const float* fc3_b  = (const float*)d_in[34];

  char* ws = (char*)d_ws;
  // Region A (41,025,536 B): S (1024 x MEMP f32) overlays {h1,h2,z}
  // (all dead before GEMM1 writes S).
  float*     h1     = (float*)(ws + 0);                    // 14,745,600 B
  float*     h2     = (float*)(ws + 14745600);             //  8,388,608 B
  float*     z      = (float*)(ws + 23134208);             //  4,194,304 B
  float*     S      = (float*)(ws + 0);                    // 41,025,536 B
  _Float16*  zn_h   = (_Float16*)(ws + 41025536);          //  2,097,152 B
  _Float16*  memT_h = (_Float16*)(ws + 43122688);          // 20,512,768 B
  _Float16*  wt_h   = (_Float16*)(ws + 63635456);          // 20,512,768 B
  _Float16*  wnt_h  = (_Float16*)(ws + 84148224);          // 20,512,768 B
  float*     z_nt   = (float*)(ws + 104660992);            //  4,194,304 B
  // Region B (25,427,968 B): mn_h (MEMP x 1024 f16 = 20,512,768 B) overlays
  // decoder/classifier scratch (mn_h dead after GEMM1).
  _Float16*  mn_h   = (_Float16*)(ws + 108855296);
  float*     dbuf1  = (float*)(ws + 108855296);            //  6,422,528 B
  float*     dbuf2  = (float*)(ws + 108855296 + 6422528);  // 12,845,056 B
  float*     c1     = (float*)(ws + 108855296 + 19267584); //  3,538,944 B
  float*     c2     = (float*)(ws + 108855296 + 22806528); //  1,048,576 B
  float*     f1     = (float*)(ws + 108855296 + 23855104); //  1,048,576 B
  float*     f2     = (float*)(ws + 108855296 + 24903680); //    524,288 B
  // Total workspace: 134,283,264 B (~128.06 MiB)

  float* out      = (float*)d_out;
  float* o_logit  = out;                // B*10
  float* o_zt     = out + 10240;        // B*1024
  float* o_rec    = out + 1058816;      // B*784
  float* o_rechat = out + 1861632;      // B*784
  float* o_wt     = out + 2664448;      // B*10000
  float* o_wnt    = out + 12904448;     // B*10000

  // --- Encoder ---
  enc1_kernel<<<14400, 256, 0, stream>>>(x, enc_w1, enc_b1, enc_g1, enc_be1, h1);
  enc2_kernel<<<8192, 256, 0, stream>>>(h1, enc_w2, enc_b2, enc_g2, enc_be2, h2);
  enc3_kernel<<<4096, 256, 0, stream>>>(h2, enc_w3, enc_b3, enc_g3, enc_be3, z);

  // --- Memory prep (independent of encoder) ---
  rownorm_f16_kernel<<<MEMP, 256, 0, stream>>>(memory, mn_h, MEM);  // pad rows zeroed
  memT_kernel<<<(FEAT * MEMP) / 256, 256, 0, stream>>>(memory, memT_h);

  // --- zn, cosine-sim logits, softmax + masking ---
  rownorm_f16_kernel<<<BATCH, 256, 0, stream>>>(z, zn_h, BATCH);
  // S = zn(1024x1024) @ mn^T(1024xMEMP): N-groups = MEMP/32 = 313, M-groups = 32
  gemm_nt_wmma2x2_kernel<<<dim3(40, 32), 256, 0, stream>>>(
      zn_h, mn_h, S, /*K=*/FEAT, /*lda=*/FEAT, /*ldb=*/FEAT, /*ldc=*/MEMP, /*ntg=*/313);
  addr_kernel<<<BATCH, 256, 0, stream>>>(S, labels, o_wt, o_wnt, wt_h, wnt_h);

  // --- z_t = w_t @ memory, z_nt = w_nt @ memory (K padded to MEMP) ---
  gemm_nt_wmma2x2_kernel<<<dim3(4, 32), 256, 0, stream>>>(
      wt_h, memT_h, o_zt, /*K=*/MEMP, /*lda=*/MEMP, /*ldb=*/MEMP, /*ldc=*/FEAT, /*ntg=*/32);
  gemm_nt_wmma2x2_kernel<<<dim3(4, 32), 256, 0, stream>>>(
      wnt_h, memT_h, z_nt, /*K=*/MEMP, /*lda=*/MEMP, /*ldb=*/MEMP, /*ldc=*/FEAT, /*ntg=*/32);

  // --- Decoder: rec_x from z_t ---
  dec1_kernel<<<6272, 256, 0, stream>>>(o_zt, dec_w1, dec_b1, dec_g1, dec_be1, dbuf1);
  dec2_kernel<<<12544, 256, 0, stream>>>(dbuf1, dec_w2, dec_b2, dec_g2, dec_be2, dbuf2);
  dec3_kernel<<<3136, 256, 0, stream>>>(dbuf2, dec_w3, dec_b3, o_rec);
  // --- Decoder: rec_x_hat from z_nt ---
  dec1_kernel<<<6272, 256, 0, stream>>>(z_nt, dec_w1, dec_b1, dec_g1, dec_be1, dbuf1);
  dec2_kernel<<<12544, 256, 0, stream>>>(dbuf1, dec_w2, dec_b2, dec_g2, dec_be2, dbuf2);
  dec3_kernel<<<3136, 256, 0, stream>>>(dbuf2, dec_w3, dec_b3, o_rechat);

  // --- Classifier on rec_x ---
  cls1_kernel<<<3456, 256, 0, stream>>>(o_rec, cls_w1, cls_b1, c1);
  cls2_kernel<<<1024, 256, 0, stream>>>(c1, cls_w2, cls_b2, c2);
  fc_kernel<<<1024, 256, 0, stream>>>(c2, fc1_w, fc1_b, f1, 256, 256, 1);
  fc_kernel<<<512, 256, 0, stream>>>(f1, fc2_w, fc2_b, f2, 256, 128, 1);
  fc_kernel<<<40, 256, 0, stream>>>(f2, fc3_w, fc3_b, o_logit, 128, 10, 0);
}
